// PointGnnLayer_50972671869115
// MI455X (gfx1250) — compile-verified, matrix operand order checked by reading, not run on hardware
//
#include <hip/hip_runtime.h>

typedef __attribute__((ext_vector_type(16))) _Float16 v16h;
typedef __attribute__((ext_vector_type(8)))  _Float16 v8h;
typedef __attribute__((ext_vector_type(4)))  _Float16 v4h;
typedef __attribute__((ext_vector_type(8)))  float    v8f;
typedef __attribute__((ext_vector_type(4)))  float    v4f;
typedef __attribute__((ext_vector_type(4)))  unsigned v4u;

#define NNODES  50000
#define NEDGES  800000
#define NT_NODE 3125     // NNODES / 16
#define WAVES   8

// ---------------------------------------------------------------- helpers

static __device__ __forceinline__ v16h cat16(v8h lo, v8h hi) {
  return __builtin_shufflevector(lo, hi, 0,1,2,3,4,5,6,7,8,9,10,11,12,13,14,15);
}

// A-fragment (16x32 f16) from an LDS row-major [16][ldk] tile, K-chunk kc.
// Per ISA layout: lanes 0-15 hold row M=lane, K = {kc*32+0..7, kc*32+16..23};
// lanes 16-31 hold the same rows with K offset +8.
static __device__ __forceinline__ v16h lds_a_frag(const _Float16* base, int ldk, int kc) {
  const int lane = threadIdx.x & 31;
  const int m    = lane & 15;
  const int k0   = kc * 32 + ((lane & 16) ? 8 : 0);
  const _Float16* p = base + m * ldk + k0;
  v8h lo = *(const v8h*)p;
  v8h hi = *(const v8h*)(p + 16);
  return cat16(lo, hi);
}

// B-fragment pre-swizzled in global memory: frag = 32 lanes x 16 halves.
static __device__ __forceinline__ v16h glb_b_frag(const _Float16* w, int frag) {
  const int lane = threadIdx.x & 31;
  const v8h* p = (const v8h*)(w + (size_t)frag * 512 + (size_t)lane * 16);
  return cat16(p[0], p[1]);
}

static __device__ __forceinline__ v8f wmma_f16(v16h a, v16h b, v8f c) {
  return __builtin_amdgcn_wmma_f32_16x16x32_f16(false, a, false, b, (short)0, c,
                                                false, false);
}

static __device__ __forceinline__ v8f bias_c(const float* b, int nt) {
  float bv = b[nt * 16 + (threadIdx.x & 15)];
  v8f c;
#pragma unroll
  for (int i = 0; i < 8; ++i) c[i] = bv;
  return c;
}

static __device__ __forceinline__ v8f relu_c(v8f c) {
#pragma unroll
  for (int i = 0; i < 8; ++i) c[i] = fmaxf(c[i], 0.0f);
  return c;
}

// C/D layout: VGPR v, lanes 0-15 -> (M=v, N=lane); lanes 16-31 -> (M=v+8).
static __device__ __forceinline__ void store_c_f16(_Float16* h, int ldk, int nt, v8f c) {
  const int lane = threadIdx.x & 31;
  const int col  = nt * 16 + (lane & 15);
  const int rb   = (lane & 16) ? 8 : 0;
#pragma unroll
  for (int v = 0; v < 8; ++v) h[(rb + v) * ldk + col] = (_Float16)c[v];
}

// Order-preserving float->u32 key (for atomicMax). Key 0 = "empty" sentinel
// (strictly below the key of every real float, including -inf).
static __device__ __forceinline__ unsigned fkey(float f) {
  unsigned b = __float_as_uint(f);
  return b ^ ((unsigned)((int)b >> 31) | 0x80000000u);
}
static __device__ __forceinline__ float dkey(unsigned u) {
  if (u == 0u) return 0.0f;                       // empty segment -> 0 (matches ref)
  unsigned b = (u & 0x80000000u) ? (u ^ 0x80000000u) : ~u;
  return __uint_as_float(b);
}

// ---------------------------------------------------------------- kernels

// Pack W [Ksrc x Nsrc] f32 into B-fragment order f16: frag(kc,nt), lane, 16 halves.
// mode 0: identity rows (zero-pad OOB). mode 1: Wf1 permutation — dst row K<64
// maps to src row K+3 (x features), K in [64,67) maps to src row K-64 (rel), else 0.
__global__ void swizzle_kernel(const float* __restrict__ W, _Float16* __restrict__ dst,
                               int Ksrc, int Nsrc, int kchunks, int ntiles, int mode) {
  const int t = blockIdx.x * blockDim.x + threadIdx.x;
  const int nfrag = kchunks * ntiles;
  if (t >= nfrag * 32) return;
  const int frag = t >> 5, lane = t & 31;
  const int kc = frag / ntiles, nt = frag % ntiles;
  _Float16* out = dst + (size_t)frag * 512 + (size_t)lane * 16;
#pragma unroll
  for (int h = 0; h < 16; ++h) {
    int K = kc * 32 + ((h < 8) ? h : h + 8) + ((lane & 16) ? 8 : 0);
    int N = nt * 16 + (lane & 15);
    int r = (mode == 1) ? ((K < 64) ? K + 3 : ((K < 67) ? K - 64 : -1)) : K;
    float v = (r >= 0 && r < Ksrc && N < Nsrc) ? W[(size_t)r * Nsrc + N] : 0.0f;
    out[h] = (_Float16)v;
  }
}

__global__ void fill_zero_kernel(unsigned* __restrict__ p, long long n) {
  long long i = (long long)blockIdx.x * blockDim.x + threadIdx.x;
  if (i < n) p[i] = 0u;
}

// mlp_h: delta = relu(x@Wh1+bh1)@Wh2+bh2 ; also persist x as f16 for edge gathers.
__global__ __launch_bounds__(256) void node_h_kernel(
    const float* __restrict__ x, const _Float16* __restrict__ wh1f,
    const float* __restrict__ bh1, const _Float16* __restrict__ wh2f,
    const float* __restrict__ bh2, float* __restrict__ delta,
    _Float16* __restrict__ xh) {
  __shared__ __attribute__((aligned(16))) _Float16 A[WAVES][16 * 64];
  __shared__ __attribute__((aligned(16))) _Float16 H[WAVES][16 * 64];
  const int lane = threadIdx.x & 31;
  const int wave = threadIdx.x >> 5;
  const int tile = blockIdx.x * WAVES + wave;
  if (tile >= NT_NODE) return;
  const int node0 = tile * 16;

  {  // x rows -> f16 LDS tile + global xh; lane pair (row, half) covers 32 floats
    const int r  = lane >> 1;
    const int hh = (lane & 1) * 32;
    const float* xr = x + (size_t)(node0 + r) * 64 + hh;
    _Float16* xg = xh + (size_t)(node0 + r) * 64 + hh;
    _Float16* al = &A[wave][r * 64 + hh];
#pragma unroll
    for (int i = 0; i < 8; ++i) {
      v4f f = ((const v4f*)xr)[i];
      v4h q;
#pragma unroll
      for (int j = 0; j < 4; ++j) q[j] = (_Float16)f[j];
      ((v4h*)al)[i] = q;
      ((v4h*)xg)[i] = q;
    }
  }
  asm volatile("s_wait_dscnt 0" ::: "memory");

#pragma unroll
  for (int nt = 0; nt < 4; ++nt) {
    v8f c = bias_c(bh1, nt);
#pragma unroll
    for (int kc = 0; kc < 2; ++kc)
      c = wmma_f16(lds_a_frag(&A[wave][0], 64, kc), glb_b_frag(wh1f, kc * 4 + nt), c);
    c = relu_c(c);
    store_c_f16(&H[wave][0], 64, nt, c);
  }
  asm volatile("s_wait_dscnt 0" ::: "memory");

  {  // layer 2: 16 cols, 3 real
    const int col = lane & 15;
    float bv = (col < 3) ? bh2[col] : 0.0f;
    v8f c;
#pragma unroll
    for (int i = 0; i < 8; ++i) c[i] = bv;
#pragma unroll
    for (int kc = 0; kc < 2; ++kc)
      c = wmma_f16(lds_a_frag(&H[wave][0], 64, kc), glb_b_frag(wh2f, kc), c);
    const int rb = (lane & 16) ? 8 : 0;
    if (col < 3) {
#pragma unroll
      for (int v = 0; v < 8; ++v)
        delta[(size_t)(node0 + rb + v) * 3 + col] = c[v];
    }
  }
}

// Fused edge MLP + segment-max (via u32-key atomicMax into L2). One wave = 16 edges.
__global__ __launch_bounds__(256) void edge_kernel(
    const float* __restrict__ pos, const long long* __restrict__ ei,
    const float* __restrict__ delta, const _Float16* __restrict__ xh,
    const _Float16* __restrict__ wf1f, const float* __restrict__ bf1,
    const _Float16* __restrict__ wf2f, const float* __restrict__ bf2,
    unsigned* __restrict__ aggk) {
  __shared__ __attribute__((aligned(16))) _Float16 A1[WAVES][16 * 96];
  __shared__ __attribute__((aligned(16))) _Float16 H[WAVES][16 * 64];
  const int lane = threadIdx.x & 31;
  const int wave = threadIdx.x >> 5;
  const long long et = ((long long)blockIdx.x * WAVES + wave) * 16;

  const long long e = et + (lane & 15);
  const int srcL = (int)ei[e];
  const int dstL = (int)ei[(long long)NEDGES + e];

  if (lane < 16) {  // rel = pos[src]-pos[dst]+delta[dst] into cols 64..66, pad 67..95
    const int s = srcL, d = dstL;
    float r0 = pos[(size_t)s * 3 + 0] - pos[(size_t)d * 3 + 0] + delta[(size_t)d * 3 + 0];
    float r1 = pos[(size_t)s * 3 + 1] - pos[(size_t)d * 3 + 1] + delta[(size_t)d * 3 + 1];
    float r2 = pos[(size_t)s * 3 + 2] - pos[(size_t)d * 3 + 2] + delta[(size_t)d * 3 + 2];
    _Float16* arow = &A1[wave][lane * 96];
    v8h z;
#pragma unroll
    for (int i = 0; i < 8; ++i) z[i] = (_Float16)0.0f;
    v8h relv = z;
    relv[0] = (_Float16)r0; relv[1] = (_Float16)r1; relv[2] = (_Float16)r2;
    *(v8h*)(arow + 64) = relv;
    *(v8h*)(arow + 72) = z;
    *(v8h*)(arow + 80) = z;
    *(v8h*)(arow + 88) = z;
  }
  {  // gather x[src] (f16, L2-resident) into cols 0..63
    const int r  = lane >> 1;
    const int hh = (lane & 1) * 32;
    const int s  = __shfl(srcL, r);
    const _Float16* xr = xh + (size_t)s * 64 + hh;
    _Float16* al = &A1[wave][r * 96 + hh];
#pragma unroll
    for (int i = 0; i < 4; ++i) ((v8h*)al)[i] = ((const v8h*)xr)[i];
  }
  asm volatile("s_wait_dscnt 0" ::: "memory");

#pragma unroll
  for (int nt = 0; nt < 4; ++nt) {  // layer 1: K=96 -> 3 WMMAs per N-tile
    v8f c = bias_c(bf1, nt);
#pragma unroll
    for (int kc = 0; kc < 3; ++kc)
      c = wmma_f16(lds_a_frag(&A1[wave][0], 96, kc), glb_b_frag(wf1f, kc * 4 + nt), c);
    c = relu_c(c);
    store_c_f16(&H[wave][0], 64, nt, c);
  }
  asm volatile("s_wait_dscnt 0" ::: "memory");

  int dsts[8];
  {
    const int rb = (lane & 16) ? 8 : 0;
#pragma unroll
    for (int v = 0; v < 8; ++v) dsts[v] = __shfl(dstL, rb + v);
  }
#pragma unroll
  for (int nt = 0; nt < 4; ++nt) {  // layer 2 + fused segment-max
    v8f c = bias_c(bf2, nt);
#pragma unroll
    for (int kc = 0; kc < 2; ++kc)
      c = wmma_f16(lds_a_frag(&H[wave][0], 64, kc), glb_b_frag(wf2f, kc * 4 + nt), c);
    const int col = nt * 16 + (lane & 15);
#pragma unroll
    for (int v = 0; v < 8; ++v)
      atomicMax(&aggk[(size_t)dsts[v] * 64 + col], fkey(c[v]));
  }
}

// mlp_g on decoded agg + residual: out = x + relu(agg@Wg1+bg1)@Wg2+bg2
__global__ __launch_bounds__(256) void node_g_kernel(
    const unsigned* __restrict__ aggk, const float* __restrict__ x,
    const _Float16* __restrict__ wg1f, const float* __restrict__ bg1,
    const _Float16* __restrict__ wg2f, const float* __restrict__ bg2,
    float* __restrict__ out) {
  __shared__ __attribute__((aligned(16))) _Float16 A[WAVES][16 * 64];
  __shared__ __attribute__((aligned(16))) _Float16 H[WAVES][16 * 64];
  const int lane = threadIdx.x & 31;
  const int wave = threadIdx.x >> 5;
  const int tile = blockIdx.x * WAVES + wave;
  if (tile >= NT_NODE) return;
  const int node0 = tile * 16;

  {  // decode u32 keys -> f16 LDS tile
    const int r  = lane >> 1;
    const int hh = (lane & 1) * 32;
    const unsigned* ar = aggk + (size_t)(node0 + r) * 64 + hh;
    _Float16* al = &A[wave][r * 64 + hh];
#pragma unroll
    for (int i = 0; i < 8; ++i) {
      v4u u = ((const v4u*)ar)[i];
      v4h q;
#pragma unroll
      for (int j = 0; j < 4; ++j) q[j] = (_Float16)dkey(u[j]);
      ((v4h*)al)[i] = q;
    }
  }
  asm volatile("s_wait_dscnt 0" ::: "memory");

#pragma unroll
  for (int nt = 0; nt < 4; ++nt) {
    v8f c = bias_c(bg1, nt);
#pragma unroll
    for (int kc = 0; kc < 2; ++kc)
      c = wmma_f16(lds_a_frag(&A[wave][0], 64, kc), glb_b_frag(wg1f, kc * 4 + nt), c);
    c = relu_c(c);
    store_c_f16(&H[wave][0], 64, nt, c);
  }
  asm volatile("s_wait_dscnt 0" ::: "memory");

#pragma unroll
  for (int nt = 0; nt < 4; ++nt) {
    v8f c = bias_c(bg2, nt);
#pragma unroll
    for (int kc = 0; kc < 2; ++kc)
      c = wmma_f16(lds_a_frag(&H[wave][0], 64, kc), glb_b_frag(wg2f, kc * 4 + nt), c);
    const int col = nt * 16 + (lane & 15);
    const int rb  = (lane & 16) ? 8 : 0;
#pragma unroll
    for (int v = 0; v < 8; ++v) {
      const size_t idx = (size_t)(node0 + rb + v) * 64 + col;
      out[idx] = x[idx] + c[v];
    }
  }
}

// ---------------------------------------------------------------- launch

extern "C" void kernel_launch(void* const* d_in, const int* in_sizes, int n_in,
                              void* d_out, int out_size, void* d_ws, size_t ws_size,
                              hipStream_t stream) {
  const float*     x   = (const float*)d_in[0];
  const float*     pos = (const float*)d_in[1];
  const long long* ei  = (const long long*)d_in[2];
  const float* Wh1 = (const float*)d_in[3];
  const float* bh1 = (const float*)d_in[4];
  const float* Wh2 = (const float*)d_in[5];
  const float* bh2 = (const float*)d_in[6];
  const float* Wf1 = (const float*)d_in[7];
  const float* bf1 = (const float*)d_in[8];
  const float* Wf2 = (const float*)d_in[9];
  const float* bf2 = (const float*)d_in[10];
  const float* Wg1 = (const float*)d_in[11];
  const float* bg1 = (const float*)d_in[12];
  const float* Wg2 = (const float*)d_in[13];
  const float* bg2 = (const float*)d_in[14];
  float* out = (float*)d_out;

  char* ws = (char*)d_ws;
  size_t off = 0;
  auto carve = [&](size_t bytes) -> void* {
    void* p = ws + off;
    off = (off + bytes + 255) & ~(size_t)255;
    return p;
  };
  _Float16* xh    = (_Float16*)carve((size_t)NNODES * 64 * 2);
  float*    delta = (float*)carve((size_t)NNODES * 3 * 4);
  unsigned* aggk  = (unsigned*)carve((size_t)NNODES * 64 * 4);
  _Float16* wh1f  = (_Float16*)carve(8 * 1024);   // 2 kc * 4 nt
  _Float16* wh2f  = (_Float16*)carve(2 * 1024);   // 2 kc * 1 nt
  _Float16* wf1f  = (_Float16*)carve(12 * 1024);  // 3 kc * 4 nt
  _Float16* wf2f  = (_Float16*)carve(8 * 1024);
  _Float16* wg1f  = (_Float16*)carve(8 * 1024);
  _Float16* wg2f  = (_Float16*)carve(8 * 1024);

  // Weight packing (tiny grids, recomputed every call for determinism)
  swizzle_kernel<<<2, 128, 0, stream>>>(Wh1, wh1f, 64, 64, 2, 4, 0);
  swizzle_kernel<<<1, 128, 0, stream>>>(Wh2, wh2f, 64, 3, 2, 1, 0);
  swizzle_kernel<<<3, 128, 0, stream>>>(Wf1, wf1f, 67, 64, 3, 4, 1);
  swizzle_kernel<<<2, 128, 0, stream>>>(Wf2, wf2f, 64, 64, 2, 4, 0);
  swizzle_kernel<<<2, 128, 0, stream>>>(Wg1, wg1f, 64, 64, 2, 4, 0);
  swizzle_kernel<<<2, 128, 0, stream>>>(Wg2, wg2f, 64, 64, 2, 4, 0);

  // agg keys -> 0 (empty sentinel)
  fill_zero_kernel<<<12500, 256, 0, stream>>>(aggk, (long long)NNODES * 64);

  // delta + f16 x
  node_h_kernel<<<(NT_NODE + WAVES - 1) / WAVES, 256, 0, stream>>>(
      x, wh1f, bh1, wh2f, bh2, delta, xh);

  // fused edge MLP + segment-max
  edge_kernel<<<NEDGES / (WAVES * 16), 256, 0, stream>>>(
      pos, ei, delta, xh, wf1f, bf1, wf2f, bf2, aggk);

  // decode + mlp_g + residual
  node_g_kernel<<<(NT_NODE + WAVES - 1) / WAVES, 256, 0, stream>>>(
      aggk, x, wg1f, bg1, wg2f, bg2, out);
}
